// biLSTM_46007689674965
// MI455X (gfx1250) — matrix operand order repaired and checked by reading, not compiled, not run on hardware
//
#include <hip/hip_runtime.h>
#include <hip/hip_bf16.h>
#include <math.h>
#include <stdint.h>

// Problem constants (from reference)
#define BSZ 256
#define SSZ 512
#define ESZ 256
#define HSZ 512
#define CSZ 10
#define TSTEPS (SSZ - 1)   // 511
#define G4H (4 * HSZ)      // 2048

typedef __bf16 bf16;
typedef __attribute__((ext_vector_type(8)))  __bf16 v8bf;
typedef __attribute__((ext_vector_type(16))) __bf16 v16bf;
typedef __attribute__((ext_vector_type(8)))  float  v8f;

// Types matching the async-to-LDS builtin signature exactly:
// param0: 'vector_size(16) int __device__ *'  (addrspace(1))
// param1: LDS-side pointer                    (addrspace(3))
typedef int v4i_gcc __attribute__((vector_size(16)));
typedef __attribute__((address_space(1))) v4i_gcc as1_v4i;
typedef __attribute__((address_space(3))) v4i_gcc as3_v4i;

#if __has_builtin(__builtin_amdgcn_global_load_async_to_lds_b128)
#define USE_ASYNC_LDS 1
#else
#define USE_ASYNC_LDS 0
#endif

__device__ __forceinline__ void wait_asynccnt0() {
#if __has_builtin(__builtin_amdgcn_s_wait_asynccnt)
    __builtin_amdgcn_s_wait_asynccnt(0);
#elif defined(__AMDGCN__)
    asm volatile("s_wait_asynccnt 0x0" ::: "memory");
#endif
}

// ---------------------------------------------------------------------------
// Small prep kernels
// ---------------------------------------------------------------------------

// dst[n*K + k] = (bf16) src[k*N + n]   (transpose + convert), writes coalesced
__global__ void transpose_bf16_kernel(bf16* __restrict__ dst,
                                      const float* __restrict__ src,
                                      int K, int N) {
    int idx = blockIdx.x * blockDim.x + threadIdx.x;
    if (idx < K * N) {
        int n = idx / K;
        int k = idx - n * K;
        dst[idx] = (bf16)src[k * N + n];
    }
}

__global__ void convert_bf16_kernel(bf16* __restrict__ dst,
                                    const float* __restrict__ src, int n) {
    int i = blockIdx.x * blockDim.x + threadIdx.x;
    if (i < n) dst[i] = (bf16)src[i];
}

// Broadcast h_init / c_init rows into state buffers (they are (1,H))
__global__ void init_state_kernel(bf16* __restrict__ hbf0_f, bf16* __restrict__ hbf0_b,
                                  float* __restrict__ c_f, float* __restrict__ c_b,
                                  const float* __restrict__ h_init,
                                  const float* __restrict__ c_init) {
    int i = blockIdx.x * blockDim.x + threadIdx.x;
    if (i < BSZ * HSZ) {
        int j = i & (HSZ - 1);
        float hv = h_init[j];
        float cv = c_init[j];
        hbf0_f[i] = (bf16)hv;
        hbf0_b[i] = (bf16)hv;
        c_f[i] = cv;
        c_b[i] = cv;
    }
}

// ---------------------------------------------------------------------------
// P = emb_bf (S x E) @ WxT^T + bias  ->  (S x 4H) fp32, both directions
// WG: 64 (S) x 128 (N) tile, 8 waves each computing 32x32 via bf16 WMMA
// ---------------------------------------------------------------------------
__global__ __launch_bounds__(256)
void pgemm_kernel(const bf16* __restrict__ embbf_f, const bf16* __restrict__ embbf_b,
                  const bf16* __restrict__ WxT_f,  const bf16* __restrict__ WxT_b,
                  const float* __restrict__ bg_f, const float* __restrict__ bi_f,
                  const float* __restrict__ bff_f, const float* __restrict__ bo_f,
                  const float* __restrict__ bg_b, const float* __restrict__ bi_b,
                  const float* __restrict__ bff_b, const float* __restrict__ bo_b,
                  float* __restrict__ P_f, float* __restrict__ P_b) {
    const int dir = blockIdx.z;
    const bf16* A  = dir ? embbf_b : embbf_f;   // (S, E)
    const bf16* BT = dir ? WxT_b  : WxT_f;      // (4H, E)  (transposed)
    float* P = dir ? P_b : P_f;                  // (S, 4H)
    const float* b0p = dir ? bg_b  : bg_f;
    const float* b1p = dir ? bi_b  : bi_f;
    const float* b2p = dir ? bff_b : bff_f;
    const float* b3p = dir ? bo_b  : bo_f;

    const int s0 = blockIdx.x * 64;
    const int n0 = blockIdx.y * 128;

    const int tid  = threadIdx.x;
    const int lane = tid & 31;
    const int wave = tid >> 5;
    const int wm = wave >> 2;        // 0..1  (M half)
    const int wn = wave & 3;         // 0..3  (N quarter)
    const int l15 = lane & 15;
    const int hi  = lane >> 4;

    __shared__ __align__(16) bf16 alds[64][32];

    v8f acc[2][2] = {};

    const int ncolbase = n0 + wn * 32;
    const bf16* bp0 = BT + (size_t)(ncolbase + l15) * ESZ;
    const bf16* bp1 = BT + (size_t)(ncolbase + 16 + l15) * ESZ;

    const int sr = tid >> 2;
    const int sc = (tid & 3) * 8;

    for (int k0 = 0; k0 < ESZ; k0 += 32) {
        __syncthreads();
        *(v8bf*)&alds[sr][sc] = *(const v8bf*)&A[(size_t)(s0 + sr) * ESZ + k0 + sc];
        __syncthreads();

        v16bf a[2];
#pragma unroll
        for (int mf = 0; mf < 2; ++mf) {
            int row = wm * 32 + mf * 16 + l15;
            v8bf lo = *(const v8bf*)&alds[row][hi * 16];
            v8bf hv = *(const v8bf*)&alds[row][hi * 16 + 8];
            a[mf] = __builtin_shufflevector(lo, hv, 0,1,2,3,4,5,6,7,8,9,10,11,12,13,14,15);
        }
        v16bf b0 = *(const v16bf*)(bp0 + k0 + hi * 16);
        v16bf b1 = *(const v16bf*)(bp1 + k0 + hi * 16);

        acc[0][0] = __builtin_amdgcn_wmma_f32_16x16x32_bf16(false, a[0], false, b0, (short)0, acc[0][0], false, false);
        acc[0][1] = __builtin_amdgcn_wmma_f32_16x16x32_bf16(false, a[0], false, b1, (short)0, acc[0][1], false, false);
        acc[1][0] = __builtin_amdgcn_wmma_f32_16x16x32_bf16(false, a[1], false, b0, (short)0, acc[1][0], false, false);
        acc[1][1] = __builtin_amdgcn_wmma_f32_16x16x32_bf16(false, a[1], false, b1, (short)0, acc[1][1], false, false);
    }

#pragma unroll
    for (int mf = 0; mf < 2; ++mf)
#pragma unroll
        for (int nf = 0; nf < 2; ++nf) {
            int n = ncolbase + nf * 16 + l15;
            int gate = n >> 9;                 // n / 512
            const float* bptr = (gate == 0) ? b0p : (gate == 1) ? b1p : (gate == 2) ? b2p : b3p;
            float bias = bptr[n & (HSZ - 1)];
#pragma unroll
            for (int r = 0; r < 8; ++r) {
                int row = s0 + wm * 32 + mf * 16 + r + 8 * hi;
                P[(size_t)row * G4H + n] = acc[mf][nf][r] + bias;
            }
        }
}

// ---------------------------------------------------------------------------
// One LSTM timestep for both directions (blockIdx.z).
// WG covers rows m0..m0+63 and hidden cols j0..j0+31 for ALL FOUR gates:
// wave w -> (gate = w&3, row-half = w>>2), 2x2 WMMA frags each (32x32 tile).
//
// The 64 x 512 bf16 h tile is CONTIGUOUS in memory -> staged into LDS once
// (via GLOBAL_LOAD_ASYNC_TO_LDS_B128 when available), then the whole K-loop
// runs barrier-free. Gate tiles are exchanged through an LDS buffer that
// overlaps (union) the h tile storage, since their lifetimes are disjoint.
// ---------------------------------------------------------------------------
union StepShMem {
    bf16  h[64][HSZ];      // 64 KB, live during the K loop
    float z[4][64][33];    // ~34 KB, live after the K loop
};

__global__ __launch_bounds__(256)
void lstm_step_kernel(const int* __restrict__ x, int t,
                      const float* __restrict__ P_f, const float* __restrict__ P_b,
                      const bf16* __restrict__ WhT_f, const bf16* __restrict__ WhT_b,
                      const bf16* __restrict__ hprev_f, const bf16* __restrict__ hprev_b,
                      bf16* __restrict__ hnext_f, bf16* __restrict__ hnext_b,
                      float* __restrict__ hf32_f, float* __restrict__ hf32_b,
                      float* __restrict__ c_f, float* __restrict__ c_b) {
    const int dir = blockIdx.z;
    const float* P    = dir ? P_b : P_f;
    const bf16* WhT   = dir ? WhT_b : WhT_f;        // (4H, H) transposed bf16
    const bf16* hprev = dir ? hprev_b : hprev_f;    // (B, H)
    bf16*  hnext = dir ? hnext_b : hnext_f;
    float* hf32  = dir ? hf32_b : hf32_f;
    float* cst   = dir ? c_b : c_f;
    const int xcol = dir ? (SSZ - 1 - t) : t;

    const int m0 = blockIdx.x * 64;
    const int j0 = blockIdx.y * 32;

    const int tid  = threadIdx.x;
    const int lane = tid & 31;
    const int wave = tid >> 5;
    const int gg = wave & 3;    // gate: 0=g 1=i 2=f 3=o
    const int mh = wave >> 2;   // which 32-row half
    const int l15 = lane & 15;
    const int hi  = lane >> 4;

    __shared__ __align__(16) StepShMem sh;

    // ---- stage the full 64x512 h tile (contiguous 64 KB) into LDS ----
    const bf16* gsrc = hprev + (size_t)m0 * HSZ;
#if USE_ASYNC_LDS
    {
        const char* g = (const char*)gsrc;
        char* l = (char*)&sh.h[0][0];
#pragma unroll
        for (int i = 0; i < 16; ++i) {
            int off = (tid + i * 256) * 16;
            __builtin_amdgcn_global_load_async_to_lds_b128(
                (as1_v4i*)(uintptr_t)(g + off),
                (as3_v4i*)(unsigned)(uintptr_t)(l + off),
                0, 0);
        }
    }
    wait_asynccnt0();
#else
#pragma unroll
    for (int i = 0; i < 16; ++i) {
        int off = (tid + i * 256) * 8;   // 8 bf16 = 16 bytes per copy
        *(v8bf*)((bf16*)&sh.h[0][0] + off) = *(const v8bf*)(gsrc + off);
    }
#endif
    __syncthreads();

    // ---- barrier-free K loop: 16 iterations x 4 WMMA ----
    v8f acc[2][2] = {};
    const bf16* bp0 = WhT + (size_t)(gg * HSZ + j0 + l15) * HSZ;
    const bf16* bp1 = WhT + (size_t)(gg * HSZ + j0 + 16 + l15) * HSZ;

    for (int k0 = 0; k0 < HSZ; k0 += 32) {
        v16bf a[2];
#pragma unroll
        for (int mf = 0; mf < 2; ++mf) {
            int row = mh * 32 + mf * 16 + l15;
            v8bf lo = *(const v8bf*)&sh.h[row][k0 + hi * 16];
            v8bf hv = *(const v8bf*)&sh.h[row][k0 + hi * 16 + 8];
            a[mf] = __builtin_shufflevector(lo, hv, 0,1,2,3,4,5,6,7,8,9,10,11,12,13,14,15);
        }
        v16bf b0 = *(const v16bf*)(bp0 + k0 + hi * 16);
        v16bf b1 = *(const v16bf*)(bp1 + k0 + hi * 16);

        acc[0][0] = __builtin_amdgcn_wmma_f32_16x16x32_bf16(false, a[0], false, b0, (short)0, acc[0][0], false, false);
        acc[0][1] = __builtin_amdgcn_wmma_f32_16x16x32_bf16(false, a[0], false, b1, (short)0, acc[0][1], false, false);
        acc[1][0] = __builtin_amdgcn_wmma_f32_16x16x32_bf16(false, a[1], false, b0, (short)0, acc[1][0], false, false);
        acc[1][1] = __builtin_amdgcn_wmma_f32_16x16x32_bf16(false, a[1], false, b1, (short)0, acc[1][1], false, false);
    }

    // h tile lifetime ends; z buffer lifetime begins (union overlap)
    __syncthreads();

    // publish this wave's gate tile to LDS
#pragma unroll
    for (int mf = 0; mf < 2; ++mf)
#pragma unroll
        for (int nf = 0; nf < 2; ++nf)
#pragma unroll
            for (int r = 0; r < 8; ++r) {
                int row = mh * 32 + mf * 16 + r + 8 * hi;
                sh.z[gg][row][nf * 16 + l15] = acc[mf][nf][r];
            }
    __syncthreads();

    // elementwise LSTM update: one row per thread, 8 columns each
    const int row = tid >> 2;
    const int cb  = (tid & 3) * 8;
    const int m = m0 + row;
    const int pr = x[(size_t)m * SSZ + xcol];
    const float* Pr = P + (size_t)pr * G4H;
#pragma unroll
    for (int q = 0; q < 8; ++q) {
        int col = cb + q;
        int j = j0 + col;
        float zg = sh.z[0][row][col] + Pr[0 * HSZ + j];
        float zi = sh.z[1][row][col] + Pr[1 * HSZ + j];
        float zf = sh.z[2][row][col] + Pr[2 * HSZ + j];
        float zo = sh.z[3][row][col] + Pr[3 * HSZ + j];
        float g  = tanhf(zg);
        float iv = 1.0f / (1.0f + __expf(-zi));
        float fv = 1.0f / (1.0f + __expf(-zf));
        float ov = 1.0f / (1.0f + __expf(-zo));
        size_t off = (size_t)m * HSZ + j;
        float cn = g * iv + cst[off] * fv;
        float hn = tanhf(cn) * ov;
        cst[off]  = cn;
        hf32[off] = hn;
        hnext[off] = (bf16)hn;
    }
}

// ---------------------------------------------------------------------------
// Final projection + log_softmax (tiny: 256 x 1024 x 10)
// ---------------------------------------------------------------------------
__global__ __launch_bounds__(256)
void final_proj_kernel(const float* __restrict__ hf, const float* __restrict__ hb,
                       const float* __restrict__ Wph, const float* __restrict__ bp,
                       float* __restrict__ out) {
    int m = blockIdx.x * blockDim.x + threadIdx.x;
    if (m >= BSZ) return;
    float logit[CSZ];
#pragma unroll
    for (int c = 0; c < CSZ; ++c) logit[c] = bp[c];
    for (int k = 0; k < HSZ; ++k) {
        float hv = hf[(size_t)m * HSZ + k];
        const float* w = Wph + (size_t)k * CSZ;
#pragma unroll
        for (int c = 0; c < CSZ; ++c) logit[c] += hv * w[c];
    }
    for (int k = 0; k < HSZ; ++k) {
        float hv = hb[(size_t)m * HSZ + k];
        const float* w = Wph + (size_t)(HSZ + k) * CSZ;
#pragma unroll
        for (int c = 0; c < CSZ; ++c) logit[c] += hv * w[c];
    }
    float mx = logit[0];
#pragma unroll
    for (int c = 1; c < CSZ; ++c) mx = fmaxf(mx, logit[c]);
    float s = 0.0f;
#pragma unroll
    for (int c = 0; c < CSZ; ++c) s += __expf(logit[c] - mx);
    float lse = mx + __logf(s);
#pragma unroll
    for (int c = 0; c < CSZ; ++c) out[(size_t)m * CSZ + c] = logit[c] - lse;
}

// ---------------------------------------------------------------------------
// Host side
// ---------------------------------------------------------------------------
extern "C" void kernel_launch(void* const* d_in, const int* in_sizes, int n_in,
                              void* d_out, int out_size, void* d_ws, size_t ws_size,
                              hipStream_t stream) {
    (void)in_sizes; (void)n_in; (void)out_size; (void)ws_size;

    const int*   x     = (const int*)d_in[0];
    const float* emb_f = (const float*)d_in[1];
    const float* Wx_f[4] = { (const float*)d_in[2],  (const float*)d_in[5],
                             (const float*)d_in[8],  (const float*)d_in[11] };
    const float* Wh_f[4] = { (const float*)d_in[3],  (const float*)d_in[6],
                             (const float*)d_in[9],  (const float*)d_in[12] };
    const float* b_f[4]  = { (const float*)d_in[4],  (const float*)d_in[7],
                             (const float*)d_in[10], (const float*)d_in[13] };
    const float* emb_b = (const float*)d_in[14];
    const float* Wx_b[4] = { (const float*)d_in[15], (const float*)d_in[18],
                             (const float*)d_in[21], (const float*)d_in[24] };
    const float* Wh_b[4] = { (const float*)d_in[16], (const float*)d_in[19],
                             (const float*)d_in[22], (const float*)d_in[25] };
    const float* b_b[4]  = { (const float*)d_in[17], (const float*)d_in[20],
                             (const float*)d_in[23], (const float*)d_in[26] };
    const float* W_ph  = (const float*)d_in[27];
    const float* b_p   = (const float*)d_in[28];
    const float* h_init = (const float*)d_in[29];
    const float* c_init = (const float*)d_in[30];
    float* out = (float*)d_out;

    // Workspace bump allocator (256B aligned)
    char* wsp = (char*)d_ws;
    auto alloc = [&](size_t bytes) -> void* {
        void* p = (void*)wsp;
        wsp += (bytes + 255) & ~(size_t)255;
        return p;
    };
    bf16* embbf_f = (bf16*)alloc((size_t)SSZ * ESZ * 2);
    bf16* embbf_b = (bf16*)alloc((size_t)SSZ * ESZ * 2);
    bf16* WxT_f   = (bf16*)alloc((size_t)G4H * ESZ * 2);
    bf16* WxT_b   = (bf16*)alloc((size_t)G4H * ESZ * 2);
    bf16* WhT_f   = (bf16*)alloc((size_t)G4H * HSZ * 2);
    bf16* WhT_b   = (bf16*)alloc((size_t)G4H * HSZ * 2);
    float* P_f    = (float*)alloc((size_t)SSZ * G4H * 4);
    float* P_b    = (float*)alloc((size_t)SSZ * G4H * 4);
    bf16* hbf_f[2] = { (bf16*)alloc((size_t)BSZ * HSZ * 2), (bf16*)alloc((size_t)BSZ * HSZ * 2) };
    bf16* hbf_b[2] = { (bf16*)alloc((size_t)BSZ * HSZ * 2), (bf16*)alloc((size_t)BSZ * HSZ * 2) };
    float* hf32_f = (float*)alloc((size_t)BSZ * HSZ * 4);
    float* hf32_b = (float*)alloc((size_t)BSZ * HSZ * 4);
    float* c_f    = (float*)alloc((size_t)BSZ * HSZ * 4);
    float* c_b    = (float*)alloc((size_t)BSZ * HSZ * 4);

    const int TB = 256;

    // Convert embeddings to bf16
    convert_bf16_kernel<<<(SSZ * ESZ + TB - 1) / TB, TB, 0, stream>>>(embbf_f, emb_f, SSZ * ESZ);
    convert_bf16_kernel<<<(SSZ * ESZ + TB - 1) / TB, TB, 0, stream>>>(embbf_b, emb_b, SSZ * ESZ);

    // Transpose+convert weights: WxT (4H x E), WhT (4H x H), gate-major rows
    for (int g = 0; g < 4; ++g) {
        transpose_bf16_kernel<<<(HSZ * ESZ + TB - 1) / TB, TB, 0, stream>>>(
            WxT_f + (size_t)g * HSZ * ESZ, Wx_f[g], ESZ, HSZ);
        transpose_bf16_kernel<<<(HSZ * ESZ + TB - 1) / TB, TB, 0, stream>>>(
            WxT_b + (size_t)g * HSZ * ESZ, Wx_b[g], ESZ, HSZ);
        transpose_bf16_kernel<<<(HSZ * HSZ + TB - 1) / TB, TB, 0, stream>>>(
            WhT_f + (size_t)g * HSZ * HSZ, Wh_f[g], HSZ, HSZ);
        transpose_bf16_kernel<<<(HSZ * HSZ + TB - 1) / TB, TB, 0, stream>>>(
            WhT_b + (size_t)g * HSZ * HSZ, Wh_b[g], HSZ, HSZ);
    }

    // Init recurrent state
    init_state_kernel<<<(BSZ * HSZ + TB - 1) / TB, TB, 0, stream>>>(
        hbf_f[0], hbf_b[0], c_f, c_b, h_init, c_init);

    // Precompute P = emb @ Wx + bias for both directions (one-time GEMM)
    pgemm_kernel<<<dim3(SSZ / 64, G4H / 128, 2), TB, 0, stream>>>(
        embbf_f, embbf_b, WxT_f, WxT_b,
        b_f[0], b_f[1], b_f[2], b_f[3],
        b_b[0], b_b[1], b_b[2], b_b[3],
        P_f, P_b);

    // Recurrent steps: 511 sequential launches, both directions per launch
    for (int t = 0; t < TSTEPS; ++t) {
        int rb = t & 1, wb = (t + 1) & 1;
        lstm_step_kernel<<<dim3(BSZ / 64, HSZ / 32, 2), TB, 0, stream>>>(
            x, t, P_f, P_b, WhT_f, WhT_b,
            hbf_f[rb], hbf_b[rb], hbf_f[wb], hbf_b[wb],
            hf32_f, hf32_b, c_f, c_b);
    }

    // Final projection + log_softmax
    final_proj_kernel<<<(BSZ + TB - 1) / TB, TB, 0, stream>>>(hf32_f, hf32_b, W_ph, b_p, out);
}
